// TemporalMemoryHub_65180423685689
// MI455X (gfx1250) — matrix operand order repaired and checked by reading, not compiled, not run on hardware
//
#include <hip/hip_runtime.h>

// ---------------- problem constants ----------------
#define DD   1024
#define NT1  256
#define NT2  128
#define NT3  64
#define NBUF 32
#define NB   16
#define TLEN 1024
#define CAT  (3*DD)          // 3072
#define COMBK (DD + NT1)     // 1280

#define NW   16              // waves per block in recurrence kernel
#define NTHR (NW*32)         // 512 threads

// ---------------- WMMA plumbing ----------------
typedef __attribute__((ext_vector_type(16))) __bf16 v16bf;
typedef __attribute__((ext_vector_type(8)))  float  v8f;

// hardware f32 -> bf16 convert (RNE), bit pattern as ushort
static __device__ __forceinline__ unsigned short f2bf(float f) {
  __bf16 h = (__bf16)f;
  return __builtin_bit_cast(unsigned short, h);
}
static __device__ __forceinline__ float bf2f(unsigned short h) {
  union { unsigned u; float f; } v; v.u = ((unsigned)h) << 16; return v.f;
}

// branch-free tanh using v_rcp (bf16-level accuracy is sufficient downstream)
static __device__ __forceinline__ float fast_tanh(float x) {
  const float ax = __builtin_fabsf(x);
  const float e  = __expf(-2.f * ax);
  const float r  = (1.f - e) * __builtin_amdgcn_rcpf(1.f + e);
  return __builtin_copysignf(r, x);
}
static __device__ __forceinline__ float sigm(float x) {
  return __builtin_amdgcn_rcpf(1.f + __expf(-x));
}

// A fragment: 16x32 (MxK), row-major bf16 source.
// ISA layout: lanes 0-15: M=lane, K={kb..kb+7, kb+16..kb+23} with kb=0;
// lanes 16-31: same rows, kb=8. Both K runs are 16B-contiguous -> two b128 loads.
static __device__ __forceinline__ v16bf load_a(const unsigned short* __restrict__ A,
                                               int lda, int lane) {
  union { v16bf v; uint4 q[2]; } f;
  const int m  = lane & 15;
  const int kb = (lane >> 4) << 3;           // 0 or 8
  const unsigned short* base = A + (size_t)m * lda + kb;
  f.q[0] = *(const uint4*)(base);            // K = kb .. kb+7
  f.q[1] = *(const uint4*)(base + 16);       // K = kb+16 .. kb+23
  return f.v;
}

// B fragment from PRE-PACKED weights: each 32x16 K-tile is a contiguous 1KB
// block; lane's 16 elements are 32 contiguous bytes -> two b128 loads.
static __device__ __forceinline__ v16bf load_bp(const unsigned short* __restrict__ PB,
                                                int lane) {
  union { v16bf v; uint4 q[2]; } f;
  const uint4* p = (const uint4*)(PB + lane * 16);
  f.q[0] = p[0];
  f.q[1] = p[1];
  return f.v;
}

static __device__ __forceinline__ v8f wmma_bf16(v16bf a, v16bf b, v8f c) {
  return __builtin_amdgcn_wmma_f32_16x16x32_bf16(false, a, false, b, (short)0, c,
                                                 false, false);
}

// loop-K GEMM tile: A row-major [16 x K] (lda), PBcol = packed B column base
static __device__ __forceinline__ v8f gemm_tile_p(const unsigned short* __restrict__ A,
                                                  int lda,
                                                  const unsigned short* __restrict__ PBcol,
                                                  int K, int lane) {
  v8f acc = {};
  for (int k = 0, blk = 0; k < K; k += 32, ++blk) {
    acc = wmma_bf16(load_a(A + k, lda, lane),
                    load_bp(PBcol + (size_t)blk * 512, lane), acc);
  }
  return acc;
}

// preloaded-A GEMM for small K reused across many n-tiles
template <int K> struct AFrags { v16bf a[K / 32]; };

template <int K>
static __device__ __forceinline__ void load_afrags(AFrags<K>& F,
                                                   const unsigned short* __restrict__ A,
                                                   int lda, int lane) {
#pragma unroll
  for (int i = 0; i < K / 32; ++i) F.a[i] = load_a(A + i * 32, lda, lane);
}

template <int K>
static __device__ __forceinline__ v8f gemm_pre(const AFrags<K>& F,
                                               const unsigned short* __restrict__ PBcol,
                                               int lane) {
  v8f acc = {};
#pragma unroll
  for (int i = 0; i < K / 32; ++i)
    acc = wmma_bf16(F.a[i], load_bp(PBcol + (size_t)i * 512, lane), acc);
  return acc;
}

// ---------------- weight f32 -> packed bf16 fragment layout ----------------
// Packed layout: block(nt, kt) at ((nt*(K/32))+kt)*512 ushorts; within a block,
// lane*16 + s where k = ((lane>>4)<<3) + (s&7) + ((s>>3)<<4), n = nt*16 + (lane&15).
__global__ void cvt_pack_kernel(const float* __restrict__ src,
                                unsigned short* __restrict__ dst,
                                int K, int N) {
  int i = blockIdx.x * 256 + threadIdx.x;
  if (i >= K * N) return;
  int k = i / N, n = i % N;
  int kt = k >> 5, nt = n >> 4;
  int lane = (((k >> 3) & 1) << 4) | (n & 15);
  int q = (((k >> 4) & 1) << 2) | ((k & 7) >> 1), r = k & 1;
  size_t blk = (size_t)nt * (K >> 5) + kt;
  dst[blk * 512 + lane * 16 + q * 2 + r] = f2bf(src[i]);
}

// ---------------- recurrence kernel (single workgroup) ----------------
struct RecParams {
  const float* x;
  const float *b_t1g, *b_t1c, *b_t1r, *b_t2v, *w_t2b_f, *b_t2b, *b_t2r;
  const float *b_t3o, *b_t3n, *b_t3r, *b_b32, *b_b21;
  const unsigned short *wt1g, *wt1c, *wt1r, *wt2v, *wt2r;
  const unsigned short *wt3o, *wt3n, *wt3r, *wb32, *wb21;
  unsigned short* R;   // [TLEN][NB][CAT] bf16
};

__launch_bounds__(NTHR, 1)
__global__ void rec_kernel(RecParams p) {
  __shared__ float          s_tier1[NB][NT1];
  __shared__ alignas(16) unsigned short s_tier1bf[NB][NT1];
  __shared__ float          s_tier2[NB][NT2];
  __shared__ alignas(16) unsigned short s_tier2bf[NB][NT2];
  __shared__ float          s_tier3[NB][NT3];
  __shared__ float          s_var3[NB][NT3];
  __shared__ alignas(16) unsigned short s_t3cat[NB][2*NT3];   // bf16 [tier3|var3]
  __shared__ alignas(16) unsigned short s_comb[NB][COMBK];    // bf16 [x_t | t1b]
  __shared__ float          s_bias21[NB][NT1];
  __shared__ float          s_bias32[NB][NT2];
  __shared__ alignas(16) unsigned short s_abuf[NB][DD];       // bf16 buffer-mean
  __shared__ float          s_z[NB][NT3];
  __shared__ float          s_r[NB][NT3];
  __shared__ float          s_beta[NB];

  const int tid  = threadIdx.x;
  const int wave = tid >> 5;
  const int lane = tid & 31;

  // ---- init state ----
  for (int i = tid; i < NB*NT1; i += NTHR) { (&s_tier1[0][0])[i] = 0.f; (&s_tier1bf[0][0])[i] = 0; }
  for (int i = tid; i < NB*NT2; i += NTHR) { (&s_tier2[0][0])[i] = 0.f; (&s_tier2bf[0][0])[i] = 0; }
  for (int i = tid; i < NB*NT3; i += NTHR) {
    (&s_tier3[0][0])[i] = 0.f; (&s_var3[0][0])[i] = 1.f;
  }
  for (int i = tid; i < NB*2*NT3; i += NTHR) {
    int n = i % (2*NT3);
    (&s_t3cat[0][0])[i] = (n < NT3) ? (unsigned short)0 : (unsigned short)0x3F80; // 0 / 1.0
  }
  __syncthreads();

  for (int t = 0; t < TLEN; ++t) {
    const int tc = t + 1;
    const bool do2 = (tc & 15) == 0;
    const bool do3 = (tc & 127) == 0;
    unsigned short* Rt = p.R + (size_t)t * NB * CAT;

    // ---- A: bias21 = tanh(tier2 @ w_b21 + b); bias32 (only needed on do2) ----
    {
      AFrags<NT2> FA;
      load_afrags<NT2>(FA, &s_tier2bf[0][0], NT2, lane);
      for (int tile = wave; tile < NT1/16; tile += NW) {
        v8f acc = gemm_pre<NT2>(FA, p.wb21 + (size_t)tile * (NT2/32) * 512, lane);
        const int n = tile*16 + (lane & 15);
        const float bb = p.b_b21[n];
#pragma unroll
        for (int i = 0; i < 8; ++i) {
          const int m = i + ((lane >> 4) << 3);
          s_bias21[m][n] = fast_tanh(acc[i] + bb);
        }
      }
    }
    if (do2) {
      AFrags<NT3> FA;
      load_afrags<NT3>(FA, &s_t3cat[0][0], 2*NT3, lane);   // tier3 only (first 64 cols)
      for (int tile = wave; tile < NT2/16; tile += NW) {
        v8f acc = gemm_pre<NT3>(FA, p.wb32 + (size_t)tile * (NT3/32) * 512, lane);
        const int n = tile*16 + (lane & 15);
        const float bb = p.b_b32[n];
#pragma unroll
        for (int i = 0; i < 8; ++i) {
          const int m = i + ((lane >> 4) << 3);
          s_bias32[m][n] = fast_tanh(acc[i] + bb);
        }
      }
    }
    __syncthreads();

    // ---- B: build comb = [x_t | tier1 + bias21] in bf16 ----
    for (int i = tid; i < NB*COMBK; i += NTHR) {
      const int m = i / COMBK, c = i % COMBK;
      float v;
      if (c < DD) v = p.x[(size_t)m * TLEN * DD + (size_t)t * DD + c];
      else        v = s_tier1[m][c - DD] + s_bias21[m][c - DD];
      s_comb[m][c] = f2bf(v);
    }
    __syncthreads();

    // ---- C: gate/cand GEMMs + tier1 update ----
    for (int tile = wave; tile < NT1/16; tile += NW) {
      const unsigned short* pg = p.wt1g + (size_t)tile * (COMBK/32) * 512;
      const unsigned short* pc = p.wt1c + (size_t)tile * (COMBK/32) * 512;
      v8f ag = {}, ac = {};
      for (int k = 0, blk = 0; k < COMBK; k += 32, ++blk) {
        v16bf a = load_a(&s_comb[0][0] + k, COMBK, lane);
        ag = wmma_bf16(a, load_bp(pg + (size_t)blk * 512, lane), ag);
        ac = wmma_bf16(a, load_bp(pc + (size_t)blk * 512, lane), ac);
      }
      const int n = tile*16 + (lane & 15);
      const float bg = p.b_t1g[n], bc = p.b_t1c[n];
#pragma unroll
      for (int i = 0; i < 8; ++i) {
        const int m = i + ((lane >> 4) << 3);
        const float g   = sigm(ag[i] + bg);
        const float c   = fast_tanh(ac[i] + bc);
        const float t1b = s_tier1[m][n] + s_bias21[m][n];
        const float nv  = g * t1b + (1.f - g) * c;
        s_tier1[m][n]   = nv;
        s_tier1bf[m][n] = f2bf(nv);
      }
    }
    __syncthreads();

    // ---- D: read1 -> R[t][:, 0:D] ----
    {
      AFrags<NT1> FA;
      load_afrags<NT1>(FA, &s_tier1bf[0][0], NT1, lane);
      for (int tile = wave; tile < DD/16; tile += NW) {
        v8f acc = gemm_pre<NT1>(FA, p.wt1r + (size_t)tile * (NT1/32) * 512, lane);
        const int n = tile*16 + (lane & 15);
        const float bb = p.b_t1r[n];
#pragma unroll
        for (int i = 0; i < 8; ++i) {
          const int m = i + ((lane >> 4) << 3);
          Rt[(size_t)m * CAT + n] = f2bf(acc[i] + bb);
        }
      }
    }

    // ---- E: tier2 delta-rule overwrite every 16 tokens ----
    if (do2) {
      const int jlo = (t - NBUF < 0) ? 0 : (t - NBUF);   // window x[jlo..t-1]
      for (int i = tid; i < NB*DD; i += NTHR) {
        const int m = i / DD, c = i % DD;
        float s = 0.f;
        for (int j = jlo; j < t; ++j) s += p.x[(size_t)m * TLEN * DD + (size_t)j * DD + c];
        s_abuf[m][c] = f2bf(s * (1.f / NBUF));
      }
      __syncthreads();
      if (tid < NB) {
        float s = 0.f;
        for (int k = 0; k < DD; ++k) s += bf2f(s_abuf[tid][k]) * p.w_t2b_f[k];
        s_beta[tid] = sigm(s + p.b_t2b[0]);
      }
      __syncthreads();
      for (int tile = wave; tile < NT2/16; tile += NW) {
        v8f acc = gemm_tile_p(&s_abuf[0][0], DD,
                              p.wt2v + (size_t)tile * (DD/32) * 512, DD, lane);
        const int n = tile*16 + (lane & 15);
        const float bv = p.b_t2v[n];
#pragma unroll
        for (int i = 0; i < 8; ++i) {
          const int m = i + ((lane >> 4) << 3);
          const float t2b = s_tier2[m][n] + s_bias32[m][n];
          const float vv  = acc[i] + bv;
          const float nv  = t2b + s_beta[m] * (vv - t2b);
          s_tier2[m][n]   = nv;
          s_tier2bf[m][n] = f2bf(nv);
        }
      }
    }
    __syncthreads();

    // ---- F: read2 -> R[t][:, D:2D] ----
    {
      AFrags<NT2> FA;
      load_afrags<NT2>(FA, &s_tier2bf[0][0], NT2, lane);
      for (int tile = wave; tile < DD/16; tile += NW) {
        v8f acc = gemm_pre<NT2>(FA, p.wt2r + (size_t)tile * (NT2/32) * 512, lane);
        const int n = tile*16 + (lane & 15);
        const float bb = p.b_t2r[n];
#pragma unroll
        for (int i = 0; i < 8; ++i) {
          const int m = i + ((lane >> 4) << 3);
          Rt[(size_t)m * CAT + DD + n] = f2bf(acc[i] + bb);
        }
      }
    }

    // ---- G: tier3 diagonal Kalman every 128 tokens (obs = read2 from R) ----
    if (do3) {
      __threadfence_block();   // flush read2 global stores before re-reading
      __syncthreads();
      for (int job = wave; job < 8; job += NW) {   // 4 z-tiles + 4 r-tiles
        const bool isz = job < 4;
        const int tt = isz ? job : job - 4;
        const unsigned short* W = isz ? p.wt3o : p.wt3n;
        v8f acc = gemm_tile_p(Rt + DD, CAT,
                              W + (size_t)tt * (DD/32) * 512, DD, lane);
        const int n = tt*16 + (lane & 15);
        const float bb = isz ? p.b_t3o[n] : p.b_t3n[n];
#pragma unroll
        for (int i = 0; i < 8; ++i) {
          const int m = i + ((lane >> 4) << 3);
          const float v = acc[i] + bb;
          if (isz) s_z[m][n] = v;
          else {
            const float sp = (v > 20.f) ? v : log1pf(__expf(v));   // softplus
            s_r[m][n] = sp + 0.01f;
          }
        }
      }
      __syncthreads();
      for (int i = tid; i < NB*NT3; i += NTHR) {
        const int m = i / NT3, n = i % NT3;
        const float sp = 0.99f * s_var3[m][n] + 0.01f;
        const float K  = sp / (sp + s_r[m][n] + 1e-8f);
        const float t3 = s_tier3[m][n] + K * (s_z[m][n] - s_tier3[m][n]);
        const float vr = (1.f - K) * sp;
        s_tier3[m][n] = t3;
        s_var3[m][n]  = vr;
        s_t3cat[m][n]       = f2bf(t3);
        s_t3cat[m][NT3 + n] = f2bf(vr);
      }
      __syncthreads();
    }

    // ---- H: read3 = [tier3|var3] @ w_t3r -> R[t][:, 2D:3D] ----
    {
      AFrags<2*NT3> FA;
      load_afrags<2*NT3>(FA, &s_t3cat[0][0], 2*NT3, lane);
      for (int tile = wave; tile < DD/16; tile += NW) {
        v8f acc = gemm_pre<2*NT3>(FA, p.wt3r + (size_t)tile * ((2*NT3)/32) * 512, lane);
        const int n = tile*16 + (lane & 15);
        const float bb = p.b_t3r[n];
#pragma unroll
        for (int i = 0; i < 8; ++i) {
          const int m = i + ((lane >> 4) << 3);
          Rt[(size_t)m * CAT + 2*DD + n] = f2bf(acc[i] + bb);
        }
      }
    }
    __syncthreads();
  }
}

// ---------------- fused GEMM: out[m,t,:] = R[t][m,:] @ w_fuse + b_fuse ----------------
__launch_bounds__(256, 1)
__global__ void fused_gemm_kernel(const unsigned short* __restrict__ R,
                                  const unsigned short* __restrict__ Wf,  // packed
                                  const float* __restrict__ bfuse,
                                  float* __restrict__ out) {
  __shared__ alignas(16) unsigned short sA[NB][CAT];   // 96 KB: 16x3072 bf16 A tile
  const int t      = blockIdx.x;           // 0..1023
  const int nstrip = blockIdx.y;           // 0..7
  const unsigned short* Ag = R + (size_t)t * NB * CAT;

  // cooperative LDS stage of A tile: CDNA5 async copy when available
  {
    const int nvec = NB * CAT * 2 / 16;    // 6144 x 16B
#if __has_builtin(__builtin_amdgcn_global_load_async_to_lds_b128)
    typedef int v4i_raw __attribute__((ext_vector_type(4)));
    typedef __attribute__((address_space(1))) v4i_raw* gv4i_p;
    typedef __attribute__((address_space(3))) v4i_raw* lv4i_p;
    const uint4* src = (const uint4*)Ag;
    uint4* dst = (uint4*)&sA[0][0];
    for (int i = threadIdx.x; i < nvec; i += 256) {
      __builtin_amdgcn_global_load_async_to_lds_b128(
          (gv4i_p)(src + i), (lv4i_p)(dst + i), 0, 0);
    }
#if __has_builtin(__builtin_amdgcn_s_wait_asynccnt)
    __builtin_amdgcn_s_wait_asynccnt(0);
#else
    asm volatile("s_wait_asynccnt 0" ::: "memory");
#endif
#else
    const uint4* src = (const uint4*)Ag;
    uint4* dst = (uint4*)&sA[0][0];
    for (int i = threadIdx.x; i < nvec; i += 256) dst[i] = src[i];
#endif
  }
  __syncthreads();

  const int wave = threadIdx.x >> 5;
  const int lane = threadIdx.x & 31;
  const int nt   = nstrip * 8 + wave;      // n-tile index, n0 = nt*16
  const unsigned short* PBcol = Wf + (size_t)nt * (CAT/32) * 512;

  v8f acc = {};
  for (int k = 0, blk = 0; k < CAT; k += 32, ++blk) {
    if (blk + 8 < CAT/32)
      __builtin_prefetch(PBcol + (size_t)(blk + 8) * 512 + lane * 16, 0, 0);
    acc = wmma_bf16(load_a(&sA[0][0] + k, CAT, lane),
                    load_bp(PBcol + (size_t)blk * 512, lane), acc);
  }

  const int n = nt * 16 + (lane & 15);
  const float bias = bfuse[n];
#pragma unroll
  for (int i = 0; i < 8; ++i) {
    const int m = i + ((lane >> 4) << 3);
    out[(size_t)m * (TLEN * DD) + (size_t)t * DD + n] = acc[i] + bias;
  }
}

// ---------------- in-place layernorm over last dim ----------------
__launch_bounds__(256, 4)
__global__ void layernorm_kernel(float* __restrict__ out,
                                 const float* __restrict__ g,
                                 const float* __restrict__ b) {
  __shared__ float rs[256], rq[256];
  float* p = out + (size_t)blockIdx.x * DD;
  float v[4]; float s = 0.f, q = 0.f;
#pragma unroll
  for (int i = 0; i < 4; ++i) {
    v[i] = p[threadIdx.x + 256 * i];
    s += v[i]; q += v[i] * v[i];
  }
  rs[threadIdx.x] = s; rq[threadIdx.x] = q;
  __syncthreads();
  for (int off = 128; off > 0; off >>= 1) {
    if (threadIdx.x < off) {
      rs[threadIdx.x] += rs[threadIdx.x + off];
      rq[threadIdx.x] += rq[threadIdx.x + off];
    }
    __syncthreads();
  }
  const float mu  = rs[0] * (1.f / DD);
  const float var = rq[0] * (1.f / DD) - mu * mu;
  const float inv = rsqrtf(var + 1e-5f);
#pragma unroll
  for (int i = 0; i < 4; ++i) {
    const int c = threadIdx.x + 256 * i;
    p[c] = (v[i] - mu) * inv * g[c] + b[c];
  }
}

// ---------------- host launcher ----------------
extern "C" void kernel_launch(void* const* d_in, const int* in_sizes, int n_in,
                              void* d_out, int out_size, void* d_ws, size_t ws_size,
                              hipStream_t stream) {
  (void)in_sizes; (void)n_in; (void)out_size; (void)ws_size;

  // workspace layout: [ R bf16 (TLEN*NB*CAT) | packed bf16 weights ]
  const size_t R_ELems = (size_t)TLEN * NB * CAT;
  unsigned short* R = (unsigned short*)d_ws;
  unsigned short* W = R + R_ELems;

  size_t off = 0;
  auto take = [&](size_t n) { unsigned short* q = W + off; off += n; return q; };
  unsigned short* wt1g  = take((size_t)COMBK * NT1);
  unsigned short* wt1c  = take((size_t)COMBK * NT1);
  unsigned short* wt1r  = take((size_t)NT1 * DD);
  unsigned short* wt2v  = take((size_t)DD * NT2);
  unsigned short* wt2r  = take((size_t)NT2 * DD);
  unsigned short* wt3o  = take((size_t)DD * NT3);
  unsigned short* wt3n  = take((size_t)DD * NT3);
  unsigned short* wt3r  = take((size_t)2 * NT3 * DD);
  unsigned short* wb32  = take((size_t)NT3 * NT2);
  unsigned short* wb21  = take((size_t)NT2 * NT1);
  unsigned short* wfuse = take((size_t)CAT * DD);

  auto cvt = [&](int idx, unsigned short* dst, int K, int N) {
    size_t n = (size_t)K * N;
    cvt_pack_kernel<<<(unsigned)((n + 255) / 256), 256, 0, stream>>>(
        (const float*)d_in[idx], dst, K, N);
  };
  cvt(1,  wt1g,  COMBK, NT1);
  cvt(3,  wt1c,  COMBK, NT1);
  cvt(5,  wt1r,  NT1, DD);
  cvt(7,  wt2v,  DD, NT2);
  cvt(11, wt2r,  NT2, DD);
  cvt(13, wt3o,  DD, NT3);
  cvt(15, wt3n,  DD, NT3);
  cvt(17, wt3r,  2*NT3, DD);
  cvt(19, wb32,  NT3, NT2);
  cvt(21, wb21,  NT2, NT1);
  cvt(23, wfuse, CAT, DD);

  RecParams p;
  p.x       = (const float*)d_in[0];
  p.b_t1g   = (const float*)d_in[2];
  p.b_t1c   = (const float*)d_in[4];
  p.b_t1r   = (const float*)d_in[6];
  p.b_t2v   = (const float*)d_in[8];
  p.w_t2b_f = (const float*)d_in[9];
  p.b_t2b   = (const float*)d_in[10];
  p.b_t2r   = (const float*)d_in[12];
  p.b_t3o   = (const float*)d_in[14];
  p.b_t3n   = (const float*)d_in[16];
  p.b_t3r   = (const float*)d_in[18];
  p.b_b32   = (const float*)d_in[20];
  p.b_b21   = (const float*)d_in[22];
  p.wt1g = wt1g; p.wt1c = wt1c; p.wt1r = wt1r;
  p.wt2v = wt2v; p.wt2r = wt2r;
  p.wt3o = wt3o; p.wt3n = wt3n; p.wt3r = wt3r;
  p.wb32 = wb32; p.wb21 = wb21;
  p.R = R;

  rec_kernel<<<1, NTHR, 0, stream>>>(p);

  fused_gemm_kernel<<<dim3(TLEN, 8), 256, 0, stream>>>(
      R, wfuse, (const float*)d_in[24], (float*)d_out);

  layernorm_kernel<<<NB * TLEN, 256, 0, stream>>>(
      (float*)d_out, (const float*)d_in[25], (const float*)d_in[26]);
}
